// DisAttention_44418551775700
// MI455X (gfx1250) — compile-verified, matrix operand order checked by reading, not run on hardware
//
#include <hip/hip_runtime.h>
#include <hip/hip_bf16.h>

// ---------------------------------------------------------------------------
// DisAttention for MI455X (gfx1250, wave32, WMMA bf16 16x16x32)
// B=4 N=1024 C=768 H=12 Dh=64
// NT store policy on the 192MB attn stream to preserve L2 for k/v reuse.
// ---------------------------------------------------------------------------

typedef __attribute__((ext_vector_type(16))) __bf16        v16bf;
typedef __attribute__((ext_vector_type(8)))  float         v8f;
typedef __attribute__((ext_vector_type(8)))  unsigned int  v8u;

#define BDIM   4
#define NDIM   1024
#define CDIM   768
#define HDIM   12
#define DHDIM  64
#define MTOT   (BDIM*NDIM)      // 4096 rows

__device__ __forceinline__ unsigned short f2bf(float f) {
    unsigned int u = __builtin_bit_cast(unsigned int, f);
    u += 0x7FFFu + ((u >> 16) & 1u);          // RNE
    return (unsigned short)(u >> 16);
}

__device__ __forceinline__ int wmma_kbase(int v, int half) {
    // 16-bit A/B fragment: VGPR v holds K-pair {kb, kb+1}
    return (v < 4) ? (8 * half + 2 * v) : (16 + 8 * half + 2 * (v - 4));
}

// Fragment whose K index is contiguous in memory (row-major [row][k]).
// Used for A tiles and for B tiles given as W[n][k] (B = W^T).
__device__ __forceinline__ v16bf load_frag_kcontig(const unsigned short* p, int ld,
                                                   int r0, int k0, int lane) {
    int half = lane >> 4, m = lane & 15;
    const unsigned short* row = p + (size_t)(r0 + m) * ld + k0;
    v8u u;
#pragma unroll
    for (int v = 0; v < 8; ++v) {
        int kb = wmma_kbase(v, half);
        u[v] = *(const unsigned int*)(row + kb);   // kb even, ld even -> aligned
    }
    return __builtin_bit_cast(v16bf, u);
}

// B fragment from a row-major B[k][n] matrix (K strided by ldb).
__device__ __forceinline__ v16bf load_b_rowmajor(const unsigned short* p, int ldb,
                                                 int k0, int n0, int lane) {
    int half = lane >> 4, n = lane & 15;
    v8u u;
#pragma unroll
    for (int v = 0; v < 8; ++v) {
        int kb = wmma_kbase(v, half);
        unsigned int lo = p[(size_t)(k0 + kb) * ldb + n0 + n];
        unsigned int hi = p[(size_t)(k0 + kb + 1) * ldb + n0 + n];
        u[v] = lo | (hi << 16);
    }
    return __builtin_bit_cast(v16bf, u);
}

// A fragment from an LDS f32 strip, with per-row scale (softmax 1/sum).
__device__ __forceinline__ v16bf load_a_lds_f32(const float* S, int ldS, int k0,
                                                int lane, const float* inv) {
    int half = lane >> 4, m = lane & 15;
    float sc = inv[m];
    const float* row = S + (size_t)m * ldS + k0;
    v8u u;
#pragma unroll
    for (int v = 0; v < 8; ++v) {
        int kb = wmma_kbase(v, half);
        unsigned int lo = f2bf(row[kb] * sc);
        unsigned int hi = f2bf(row[kb + 1] * sc);
        u[v] = lo | (hi << 16);
    }
    return __builtin_bit_cast(v16bf, u);
}

__device__ __forceinline__ v8f wmma_bf16(v16bf a, v16bf b, v8f c) {
    return __builtin_amdgcn_wmma_f32_16x16x32_bf16(false, a, false, b,
                                                   (short)0, c, false, false);
}

// ---------------------------------------------------------------------------
// K0: fp32 -> bf16 (raw u16) conversion
// ---------------------------------------------------------------------------
__global__ void cvt_bf16_kernel(const float* __restrict__ src,
                                unsigned short* __restrict__ dst, int n) {
    int i = blockIdx.x * blockDim.x + threadIdx.x;
    if (i < n) dst[i] = f2bf(src[i]);
}

// ---------------------------------------------------------------------------
// K1: qkv = x @ qkv_w^T, scattered into q/k/v buffers [B][H][N][Dh] (bf16)
// One wave = one 16x16 output tile. 256 x 144 tiles.
// ---------------------------------------------------------------------------
__global__ void qkv_gemm_kernel(const unsigned short* __restrict__ xb,
                                const unsigned short* __restrict__ qwb,
                                unsigned short* __restrict__ qb,
                                unsigned short* __restrict__ kb,
                                unsigned short* __restrict__ vb) {
    int wave = (blockIdx.x * blockDim.x + threadIdx.x) >> 5;
    int lane = threadIdx.x & 31;
    int mt = wave / 144, nt = wave % 144;
    int m0 = mt * 16, n0 = nt * 16;

    v8f c = {};
#pragma unroll 4
    for (int k0 = 0; k0 < CDIM; k0 += 32) {
        v16bf a = load_frag_kcontig(xb,  CDIM, m0, k0, lane);
        v16bf b = load_frag_kcontig(qwb, CDIM, n0, k0, lane);
        c = wmma_bf16(a, b, c);
    }

    int half = lane >> 4, col = lane & 15;
    int n = n0 + col;
    int which = n / CDIM, rem = n % CDIM;
    int h = rem >> 6, d = rem & 63;
    unsigned short* dst = (which == 0) ? qb : ((which == 1) ? kb : vb);
#pragma unroll
    for (int r = 0; r < 8; ++r) {
        int m = m0 + r + 8 * half;
        int bidx = m >> 10, nn = m & 1023;
        dst[((((size_t)bidx * HDIM + h) * NDIM + nn) << 6) + d] = f2bf(c[r]);
    }
}

// ---------------------------------------------------------------------------
// K2: fused attention. One block (8 waves) per (b, h, 16-row tile of n).
//   - scores(16x1024) = q_tile . k^T * scale + mask      (WMMA, into LDS)
//   - softmax + weight renorm: final = p*w / sum(p*w)
//   - write attn tile to d_out (non-temporal: 192MB stream, never re-read)
//   - out_tile(16x64) = attn_tile @ v                    (WMMA, A from LDS)
// Dynamic LDS: 16*1024 f32 scores + 2*16*64 f32 partials + 16 f32 inv
// ---------------------------------------------------------------------------
__global__ void attn_fused_kernel(const unsigned short* __restrict__ qb,
                                  const unsigned short* __restrict__ kb,
                                  const unsigned short* __restrict__ vb,
                                  const float* __restrict__ mask,
                                  const float* __restrict__ weight,
                                  float* __restrict__ attn_out,
                                  unsigned short* __restrict__ outb) {
    extern __shared__ float smem[];
    float* S   = smem;                    // 16*1024
    float* P   = smem + 16 * 1024;        // 2*16*64
    float* INV = P + 2 * 16 * 64;         // 16

    int blk = blockIdx.x;                 // 48*64 blocks
    int nt = blk & 63;
    int bh = blk >> 6;
    int h  = bh % HDIM;
    int b  = bh / HDIM;
    int n0 = nt * 16;

    int wave = threadIdx.x >> 5, lane = threadIdx.x & 31;
    int half = lane >> 4, col = lane & 15;

    const unsigned short* qbase = qb + (size_t)bh * NDIM * DHDIM;
    const unsigned short* kbase = kb + (size_t)bh * NDIM * DHDIM;
    const unsigned short* vbase = vb + (size_t)bh * NDIM * DHDIM;

    const float scale = rsqrtf((float)(CDIM * HDIM));   // (C*H)^-0.5

    // ---- scores: each wave covers 8 of 64 key tiles ----
    v16bf a0 = load_frag_kcontig(qbase, DHDIM, n0, 0,  lane);
    v16bf a1 = load_frag_kcontig(qbase, DHDIM, n0, 32, lane);
#pragma unroll
    for (int i = 0; i < 8; ++i) {
        int mt = i * 8 + wave;
        v16bf b0 = load_frag_kcontig(kbase, DHDIM, mt * 16, 0,  lane);
        v16bf b1 = load_frag_kcontig(kbase, DHDIM, mt * 16, 32, lane);
        v8f c = {};
        c = wmma_bf16(a0, b0, c);
        c = wmma_bf16(a1, b1, c);
#pragma unroll
        for (int r = 0; r < 8; ++r) {
            int row = r + 8 * half;
            int gm  = mt * 16 + col;
            // mask is read once per head (12x): NT load keeps it out of L2
            float mk = __builtin_nontemporal_load(
                &mask[((size_t)b * NDIM + n0 + row) * NDIM + gm]);
            S[row * 1024 + gm] = c[r] * scale + mk;
        }
    }
    __syncthreads();

    // ---- softmax + weight renorm: each wave owns 2 rows ----
#pragma unroll
    for (int rr = 0; rr < 2; ++rr) {
        int row = wave * 2 + rr;
        float* sr = S + row * 1024;
        float mx = -1e30f;
        for (int j = lane; j < 1024; j += 32) mx = fmaxf(mx, sr[j]);
#pragma unroll
        for (int o = 16; o > 0; o >>= 1) mx = fmaxf(mx, __shfl_xor(mx, o, 32));
        float sum = 0.f;
        for (int j = lane; j < 1024; j += 32) {
            float p  = __expf(sr[j] - mx);
            float w  = weight[b * NDIM + j] + 1e-10f;
            float pw = p * w;
            sr[j] = pw;
            sum  += pw;
        }
#pragma unroll
        for (int o = 16; o > 0; o >>= 1) sum += __shfl_xor(sum, o, 32);
        if (lane == 0) INV[row] = 1.0f / sum;
    }
    __syncthreads();

    // ---- write normalized attention tile (streaming NT stores) ----
    {
        float* arow = attn_out + ((size_t)bh * NDIM + n0) * NDIM;
        for (int idx = threadIdx.x; idx < 16 * 1024; idx += 256) {
            int row = idx >> 10, j = idx & 1023;
            __builtin_nontemporal_store(S[row * 1024 + j] * INV[row],
                                        &arow[(size_t)row * NDIM + j]);
        }
    }
    __syncthreads();

    // ---- out tile = attn @ v : wave -> (dtile = wave&3, khalf = wave>>2) ----
    int dtile = wave & 3, khalf = wave >> 2;
    v8f c = {};
    for (int kk = 0; kk < 16; ++kk) {
        int k0 = (khalf * 16 + kk) * 32;
        v16bf a  = load_a_lds_f32(S, 1024, k0, lane, INV);
        v16bf bf = load_b_rowmajor(vbase, DHDIM, k0, dtile * 16, lane);
        c = wmma_bf16(a, bf, c);
    }
#pragma unroll
    for (int r = 0; r < 8; ++r) {
        int row = r + 8 * half;
        P[khalf * 1024 + row * 64 + dtile * 16 + col] = c[r];
    }
    __syncthreads();

    for (int idx = threadIdx.x; idx < 1024; idx += 256) {
        int row = idx >> 6, d = idx & 63;
        float v2 = P[idx] + P[1024 + idx];
        outb[((size_t)b * NDIM + n0 + row) * CDIM + h * DHDIM + d] = f2bf(v2);
    }
}

// ---------------------------------------------------------------------------
// K3: mu = out[:,:,:384] @ mu_w^T + mu_b ; logsigma = out[:,:,384:] @ ls_w^T + ls_b
// One wave = one 16x16 tile; tiles = 2(sel) x 256(m) x 48(j)
// ---------------------------------------------------------------------------
__global__ void head_gemm_kernel(const unsigned short* __restrict__ outb,
                                 const unsigned short* __restrict__ muwb,
                                 const unsigned short* __restrict__ lswb,
                                 const float* __restrict__ mu_bias,
                                 const float* __restrict__ ls_bias,
                                 float* __restrict__ mu,
                                 float* __restrict__ ls) {
    int wave = (blockIdx.x * blockDim.x + threadIdx.x) >> 5;
    int lane = threadIdx.x & 31;
    int jt   = wave % 48;
    int rest = wave / 48;
    int mt   = rest % 256;
    int sel  = rest / 256;

    const unsigned short* W = sel ? lswb : muwb;
    const float* bias = sel ? ls_bias : mu_bias;
    float* dst = sel ? ls : mu;
    int m0 = mt * 16, j0 = jt * 16;
    int koff = sel ? (CDIM / 2) : 0;

    v8f c = {};
#pragma unroll 4
    for (int k0 = 0; k0 < CDIM / 2; k0 += 32) {
        v16bf a = load_frag_kcontig(outb, CDIM, m0, koff + k0, lane);
        v16bf b = load_frag_kcontig(W, CDIM / 2, j0, k0, lane);
        c = wmma_bf16(a, b, c);
    }

    int half = lane >> 4, col = lane & 15;
    float bj = bias[j0 + col];
#pragma unroll
    for (int r = 0; r < 8; ++r) {
        int m = m0 + r + 8 * half;
        __builtin_nontemporal_store(c[r] + bj,
                                    &dst[(size_t)m * CDIM + j0 + col]);
    }
}

// ---------------------------------------------------------------------------
extern "C" void kernel_launch(void* const* d_in, const int* in_sizes, int n_in,
                              void* d_out, int out_size, void* d_ws, size_t ws_size,
                              hipStream_t stream) {
    const float* x      = (const float*)d_in[0];   // 4*1024*768
    const float* mask   = (const float*)d_in[1];   // 4*1*1024*1024
    const float* weight = (const float*)d_in[2];   // 4*1024
    const float* qkv_w  = (const float*)d_in[3];   // 2304*768
    const float* mu_w   = (const float*)d_in[4];   // 768*384
    const float* mu_b   = (const float*)d_in[5];   // 768
    const float* ls_w   = (const float*)d_in[6];   // 768*384
    const float* ls_b   = (const float*)d_in[7];   // 768

    float* mu   = (float*)d_out;
    float* ls   = mu + (size_t)MTOT * CDIM;
    float* attn = ls + (size_t)MTOT * CDIM;

    const size_t NX   = (size_t)MTOT * CDIM;        // 3,145,728
    const size_t NQW  = (size_t)3 * CDIM * CDIM;    // 1,769,472
    const size_t NHW  = (size_t)CDIM * (CDIM / 2);  //   294,912

    unsigned short* ws   = (unsigned short*)d_ws;
    unsigned short* xb   = ws;
    unsigned short* qwb  = xb   + NX;
    unsigned short* muwb = qwb  + NQW;
    unsigned short* lswb = muwb + NHW;
    unsigned short* qb   = lswb + NHW;
    unsigned short* kb   = qb   + NX;
    unsigned short* vb   = kb   + NX;
    unsigned short* outb = vb   + NX;

    // K0: conversions
    cvt_bf16_kernel<<<(int)((NX  + 255) / 256), 256, 0, stream>>>(x,     xb,   (int)NX);
    cvt_bf16_kernel<<<(int)((NQW + 255) / 256), 256, 0, stream>>>(qkv_w, qwb,  (int)NQW);
    cvt_bf16_kernel<<<(int)((NHW + 255) / 256), 256, 0, stream>>>(mu_w,  muwb, (int)NHW);
    cvt_bf16_kernel<<<(int)((NHW + 255) / 256), 256, 0, stream>>>(ls_w,  lswb, (int)NHW);

    // K1: QKV projection (256 x 144 tiles, 8 waves / block)
    qkv_gemm_kernel<<<(256 * 144) / 8, 256, 0, stream>>>(xb, qwb, qb, kb, vb);

    // K2: fused attention (48 bh * 64 n-tiles blocks; 73,792 B dynamic LDS)
    size_t lds = (16 * 1024 + 2 * 16 * 64 + 16) * sizeof(float);
    attn_fused_kernel<<<48 * 64, 256, lds, stream>>>(qb, kb, vb, mask, weight,
                                                     attn, outb);

    // K3: mu / logsigma projections (2*256*48 tiles, 8 waves / block)
    head_gemm_kernel<<<(2 * 256 * 48) / 8, 256, 0, stream>>>(outb, muwb, lswb,
                                                             mu_b, ls_b, mu, ls);
}
